// LSTMSeq2Seq_74002286510352
// MI455X (gfx1250) — matrix operand order repaired and checked by reading, not compiled, not run on hardware
//
#include <hip/hip_runtime.h>

// ---------------------------------------------------------------------------
// CDNA5 (gfx1250) seq2seq LSTM + attention, WMMA f16 (16x16x32), wave32.
// ---------------------------------------------------------------------------

typedef _Float16 half_t;
typedef __attribute__((ext_vector_type(16))) _Float16 v16h;
typedef __attribute__((ext_vector_type(8)))  float    v8f;

#define NB   4096   // batch
#define LS   64     // encoder seq len
#define TS   48     // decoder seq len
#define HH   64     // hidden

__device__ inline v8f wmma16(v16h a, v16h b, v8f c) {
  return __builtin_amdgcn_wmma_f32_16x16x32_f16(false, a, false, b, (short)0, c,
                                                false, false);
}

// Fast hardware reciprocal (v_rcp_f32): avoids the IEEE div_scale/div_fmas
// sequence on the sequential LSTM critical path.
__device__ inline float rcpf(float x) { return __builtin_amdgcn_rcpf(x); }

// A fragment (16x32, f16). Row-major source with leading dim ldm (halves).
// Layout: lanes 0-15 -> K 0..15, lanes 16-31 -> K 16..31.
__device__ inline v16h a_frag_f16(const half_t* base, int ldm, int lane, int kb) {
  int row  = lane & 15;
  int koff = (lane & 16) ? 16 : 0;
  const half_t* p = base + row * ldm + kb + koff;
  v16h a;
#pragma unroll
  for (int e = 0; e < 8; ++e) { a[e] = p[e]; a[8 + e] = p[8 + e]; }
  return a;
}

// A fragment with only K = 0..15 valid (K 16..31 zero-padded): for K-dim = 16 GEMMs.
__device__ inline v16h a_frag_k16(const half_t* base, int ldm, int lane) {
  int row  = lane & 15;
  int koff = (lane & 16) ? 8 : 0;
  const half_t* p = base + row * ldm + koff;
  v16h a;
#pragma unroll
  for (int e = 0; e < 8; ++e) { a[e] = (lane & 16) ? (half_t)0.f : p[e]; a[8 + e] = (lane & 16) ? (half_t)0.f : p[8 + e]; }
  return a;
}

// B fragment (32x16) from f32 weight W[N][ldw] row-major: B(k,n) = W[n][k].
// Layout: lanes 0-15 hold K kb..kb+15, lanes 16-31 hold K kb+16..kb+31; col = lane&15.
__device__ inline v16h b_frag_w32(const float* W, int ldw, int ncol, int lane, int kb) {
  int koff = (lane & 16) ? 16 : 0;
  const float* p = W + ncol * ldw + kb + koff;
  v16h b;
#pragma unroll
  for (int e = 0; e < 16; ++e) b[e] = (half_t)p[e];
  return b;
}

// B fragment from f16 matrix stored [K][ldn] (row index is K).
__device__ inline v16h b_frag_kxn(const half_t* Bm, int ldn, int ncol, int lane, int kb) {
  int koff = (lane & 16) ? 16 : 0;
  v16h b;
#pragma unroll
  for (int e = 0; e < 16; ++e) b[e] = Bm[(kb + koff + e) * ldn + ncol];
  return b;
}

// B fragment for scores GEMM: B(k=d, n=key) = Kmat[key][d], only K 0..15 real.
__device__ inline v16h b_frag_k16(const half_t* Kmat, int ncol, int lane) {
  v16h b;
  if (lane < 16) {
#pragma unroll
    for (int e = 0; e < 16; ++e) b[e] = Kmat[ncol * 16 + e];
  } else {
#pragma unroll
    for (int e = 0; e < 16; ++e) b[e] = (half_t)0.f;
  }
  return b;
}

// Branch-free sigmoid / tanh via hardware v_exp_f32 + v_rcp_f32.
// tanh saturates to +/-1 at the extremes (exp -> inf or 0), no NaNs.
__device__ inline float sigf(float x) { return rcpf(1.f + __expf(-x)); }
__device__ inline float tanh_fast(float x) {
  return 1.f - 2.f * rcpf(__expf(2.f * x) + 1.f);
}

// ---------------------------------------------------------------------------
// Fused LSTM scan: per block 16 batch rows, 16 waves = 16 gate N-tiles.
// Per step each wave: g-tile = x_t@Wih^T + h@Whh^T  (4 WMMAs, K=64 in 2 chunks).
// mode 0: x f32 [B,S,64]; mode 1: x f16 [B,S,64];
// mode 2: t==0 from x0 f32 [B,64], else xf32 (target) at t-1.
// ---------------------------------------------------------------------------
__global__ __launch_bounds__(512) void lstm_scan_kernel(
    int mode, int S,
    const float* __restrict__ xf32, const half_t* __restrict__ xf16,
    const float* __restrict__ x0,
    const float* __restrict__ Wih, const float* __restrict__ Whh,
    const float* __restrict__ bih, const float* __restrict__ bhh,
    const float* __restrict__ h0, const float* __restrict__ c0,
    half_t* __restrict__ hseq, float* __restrict__ hT, float* __restrict__ cT)
{
  __shared__ half_t x_lds[16 * 72];
  __shared__ half_t h_lds[16 * 72];
  __shared__ float  g_lds[16 * 260];

  const int tid  = threadIdx.x;
  const int wave = tid >> 5;
  const int lane = tid & 31;
  const int b0   = blockIdx.x * 16;

  // gate-combine mapping: thread owns pairs (prow, pu) and (prow, pu+1)
  const int prow = (2 * tid) >> 6;
  const int pu   = (2 * tid) & 63;

  // Resident B-fragments for this wave's 16 gate columns (held across all steps)
  const int ncol = wave * 16 + (lane & 15);          // 0..255
  const v16h wxB0 = b_frag_w32(Wih, HH, ncol, lane, 0);
  const v16h wxB1 = b_frag_w32(Wih, HH, ncol, lane, 32);
  const v16h whB0 = b_frag_w32(Whh, HH, ncol, lane, 0);
  const v16h whB1 = b_frag_w32(Whh, HH, ncol, lane, 32);

  float cl[2], hl[2];
#pragma unroll
  for (int k = 0; k < 2; ++k) {
    int u = pu + k;
    float hv = h0 ? h0[(b0 + prow) * HH + u] : 0.f;
    float cv = c0 ? c0[(b0 + prow) * HH + u] : 0.f;
    h_lds[prow * 72 + u] = (half_t)hv;
    hl[k] = hv; cl[k] = cv;
  }

  const int r0   = (lane & 16) ? 8 : 0;
  const int gcol = wave * 16 + (lane & 15);

  for (int t = 0; t < S; ++t) {
    // stage x_t tile as f16
#pragma unroll
    for (int k = 0; k < 2; ++k) {
      int u = pu + k;
      float xv;
      if (mode == 0)      xv = xf32[((b0 + prow) * S + t) * HH + u];
      else if (mode == 1) xv = (float)xf16[((b0 + prow) * S + t) * HH + u];
      else                xv = (t == 0) ? x0[(b0 + prow) * HH + u]
                                        : xf32[((b0 + prow) * S + (t - 1)) * HH + u];
      x_lds[prow * 72 + u] = (half_t)xv;
    }
    __syncthreads();

    v8f acc = {0.f, 0.f, 0.f, 0.f, 0.f, 0.f, 0.f, 0.f};
    acc = wmma16(a_frag_f16(x_lds, 72, lane, 0),  wxB0, acc);
    acc = wmma16(a_frag_f16(x_lds, 72, lane, 32), wxB1, acc);
    acc = wmma16(a_frag_f16(h_lds, 72, lane, 0),  whB0, acc);
    acc = wmma16(a_frag_f16(h_lds, 72, lane, 32), whB1, acc);
#pragma unroll
    for (int v = 0; v < 8; ++v) g_lds[(r0 + v) * 260 + gcol] = acc[v];
    __syncthreads();

    // gate combine (PyTorch order i, f, g, o)
#pragma unroll
    for (int k = 0; k < 2; ++k) {
      int u = pu + k;
      float gi = g_lds[prow * 260 + u]       + bih[u]       + bhh[u];
      float gf = g_lds[prow * 260 + 64 + u]  + bih[64 + u]  + bhh[64 + u];
      float gg = g_lds[prow * 260 + 128 + u] + bih[128 + u] + bhh[128 + u];
      float go = g_lds[prow * 260 + 192 + u] + bih[192 + u] + bhh[192 + u];
      float cn = sigf(gf) * cl[k] + sigf(gi) * tanh_fast(gg);
      float hn = sigf(go) * tanh_fast(cn);
      cl[k] = cn; hl[k] = hn;
      h_lds[prow * 72 + u] = (half_t)hn;
      hseq[((b0 + prow) * S + t) * HH + u] = (half_t)hn;
    }
    __syncthreads();
  }

#pragma unroll
  for (int k = 0; k < 2; ++k) {
    int u = pu + k;
    hT[(b0 + prow) * HH + u] = hl[k];
    cT[(b0 + prow) * HH + u] = cl[k];
  }
}

// ---------------------------------------------------------------------------
// K/V projection: kv16[B*L, 128] = enc_out_f16 @ [Wk;Wv]^T + [bk;bv]  (f16 out)
// One block = 16 M rows, wave w = N-tile w (N = 128 exactly = 8 tiles).
// ---------------------------------------------------------------------------
__global__ __launch_bounds__(256) void kv_proj_kernel(
    const half_t* __restrict__ enc16, const float* __restrict__ Wqkv,
    const float* __restrict__ bqkv, half_t* __restrict__ kv16)
{
  const int tid = threadIdx.x, wave = tid >> 5, lane = tid & 31;
  const int m0 = blockIdx.x * 16;
  const int nc = wave * 16 + (lane & 15);            // 0..127
  v8f acc = {0.f, 0.f, 0.f, 0.f, 0.f, 0.f, 0.f, 0.f};
  acc = wmma16(a_frag_f16(enc16 + m0 * HH, HH, lane, 0),
               b_frag_w32(Wqkv + HH * HH, HH, HH + nc, lane, 0), acc);
  acc = wmma16(a_frag_f16(enc16 + m0 * HH, HH, lane, 32),
               b_frag_w32(Wqkv + HH * HH, HH, HH + nc, lane, 32), acc);
  const float bb = bqkv[HH + nc];
  const int r0 = (lane & 16) ? 8 : 0;
#pragma unroll
  for (int v = 0; v < 8; ++v)
    kv16[(m0 + r0 + v) * 128 + nc] = (half_t)(acc[v] + bb);
}

// ---------------------------------------------------------------------------
// Step-0 attention: q = enc_out[:, L-1] @ Wq^T + bq, attend over K/V, project Wo.
// One block per b; wave = head; scalar dot products (M=1) + shuffle softmax.
// ---------------------------------------------------------------------------
__global__ __launch_bounds__(128) void attn0_kernel(
    const half_t* __restrict__ enc16, const half_t* __restrict__ kv16,
    const float* __restrict__ Wqkv, const float* __restrict__ bqkv,
    const float* __restrict__ Wo, const float* __restrict__ bo,
    float* __restrict__ dec_in0)
{
  __shared__ float qbuf[4][16];
  __shared__ float wbuf[4][64];
  __shared__ float obuf[64];
  const int tid = threadIdx.x, h = tid >> 5, lane = tid & 31;
  const int b = blockIdx.x;

  if (lane < 16) {
    const half_t* e = enc16 + (b * LS + (LS - 1)) * HH;
    const float*  wq = Wqkv + (h * 16 + lane) * HH;
    float acc = bqkv[h * 16 + lane];
    for (int k = 0; k < HH; ++k) acc += (float)e[k] * wq[k];
    qbuf[h][lane] = acc;
  }
  __syncthreads();

  float s0 = 0.f, s1 = 0.f;
  {
    const half_t* k0 = kv16 + (b * LS + lane) * 128 + h * 16;
    const half_t* k1 = kv16 + (b * LS + lane + 32) * 128 + h * 16;
    for (int d = 0; d < 16; ++d) {
      s0 += qbuf[h][d] * (float)k0[d];
      s1 += qbuf[h][d] * (float)k1[d];
    }
    s0 *= 0.25f; s1 *= 0.25f;
  }
  float m = fmaxf(s0, s1);
  for (int off = 16; off > 0; off >>= 1) m = fmaxf(m, __shfl_xor(m, off, 32));
  float e0 = __expf(s0 - m), e1 = __expf(s1 - m);
  float sum = e0 + e1;
  for (int off = 16; off > 0; off >>= 1) sum += __shfl_xor(sum, off, 32);
  float inv = rcpf(sum);
  wbuf[h][lane] = e0 * inv;
  wbuf[h][lane + 32] = e1 * inv;
  __syncthreads();

  if (lane < 16) {
    float acc = 0.f;
    for (int j = 0; j < LS; ++j)
      acc += wbuf[h][j] * (float)kv16[(b * LS + j) * 128 + 64 + h * 16 + lane];
    obuf[h * 16 + lane] = acc;
  }
  __syncthreads();

  if (tid < HH) {
    float acc = bo[tid];
    for (int k = 0; k < HH; ++k) acc += obuf[k] * Wo[tid * HH + k];
    dec_in0[b * HH + tid] = acc;
  }
}

// ---------------------------------------------------------------------------
// Cross attention + Wo + FC head, fused. One block per batch sample b.
// Q-proj, per-head scores (K=16 zero-padded WMMA), softmax (+head-mean weights
// straight to d_out), W@V, Wo projection, FC1 relu (WMMA), FC2 (N=3, scalar).
// ---------------------------------------------------------------------------
__global__ __launch_bounds__(256) void cross_attn_kernel(
    const half_t* __restrict__ dec16, const half_t* __restrict__ kv16,
    const float* __restrict__ Wqkv, const float* __restrict__ bqkv,
    const float* __restrict__ Wo, const float* __restrict__ bo,
    const float* __restrict__ W1, const float* __restrict__ b1,
    const float* __restrict__ W2, const float* __restrict__ b2,
    float* __restrict__ pred, float* __restrict__ attnw)
{
  __shared__ half_t q16[TS * 64];
  __shared__ half_t kK[64 * 16];
  __shared__ half_t kV[64 * 16];
  __shared__ float  sc[TS * 64];
  __shared__ half_t w16[TS * 64];
  __shared__ float  wsum[TS * 64];
  __shared__ half_t ao16[TS * 64];
  __shared__ half_t po16[TS * 64];
  __shared__ half_t h16[TS * 32];

  const int tid = threadIdx.x, wave = tid >> 5, lane = tid & 31;
  const int b = blockIdx.x;
  const int r0 = (lane & 16) ? 8 : 0;
  const half_t* db = dec16 + b * TS * HH;

  for (int i = tid; i < TS * 64; i += 256) wsum[i] = 0.f;

  // Q projection: (48x64) @ Wq^T + bq -> q16 (f16)
  for (int job = wave; job < 12; job += 8) {
    int mt = job >> 2, nt = job & 3;
    int nc = nt * 16 + (lane & 15);
    v8f c = {0.f, 0.f, 0.f, 0.f, 0.f, 0.f, 0.f, 0.f};
    c = wmma16(a_frag_f16(db + mt * 16 * HH, HH, lane, 0),
               b_frag_w32(Wqkv, HH, nc, lane, 0), c);
    c = wmma16(a_frag_f16(db + mt * 16 * HH, HH, lane, 32),
               b_frag_w32(Wqkv, HH, nc, lane, 32), c);
    float bb = bqkv[nc];
#pragma unroll
    for (int v = 0; v < 8; ++v)
      q16[(mt * 16 + r0 + v) * 64 + nc] = (half_t)(c[v] + bb);
  }
  __syncthreads();

  for (int h = 0; h < 4; ++h) {
    for (int i = tid; i < 64 * 16; i += 256) {
      int j = i >> 4, d = i & 15;
      kK[i] = kv16[(b * LS + j) * 128 + h * 16 + d];
      kV[i] = kv16[(b * LS + j) * 128 + 64 + h * 16 + d];
    }
    __syncthreads();

    // scores = Q_h (48x16) @ K^T (16x64), scale 1/sqrt(16)
    for (int job = wave; job < 12; job += 8) {
      int mt = job >> 2, nt = job & 3;
      int nc = nt * 16 + (lane & 15);
      v8f c = {0.f, 0.f, 0.f, 0.f, 0.f, 0.f, 0.f, 0.f};
      c = wmma16(a_frag_k16(q16 + mt * 16 * 64 + h * 16, 64, lane),
                 b_frag_k16(kK, nc, lane), c);
#pragma unroll
      for (int v = 0; v < 8; ++v)
        sc[(mt * 16 + r0 + v) * 64 + nc] = 0.25f * c[v];
    }
    __syncthreads();

    if (tid < TS) {
      float m = -3.0e38f;
      for (int j = 0; j < 64; ++j) m = fmaxf(m, sc[tid * 64 + j]);
      float s = 0.f;
      for (int j = 0; j < 64; ++j) {
        float e = __expf(sc[tid * 64 + j] - m);
        sc[tid * 64 + j] = e; s += e;
      }
      float inv = rcpf(s);
      for (int j = 0; j < 64; ++j) {
        float w = sc[tid * 64 + j] * inv;
        w16[tid * 64 + j]  = (half_t)w;
        wsum[tid * 64 + j] += 0.25f * w;   // mean over 4 heads
      }
    }
    __syncthreads();

    // out_h = W (48x64) @ V (64x16)
    for (int job = wave; job < 3; job += 8) {
      v8f c = {0.f, 0.f, 0.f, 0.f, 0.f, 0.f, 0.f, 0.f};
      c = wmma16(a_frag_f16(w16 + job * 16 * 64, 64, lane, 0),
                 b_frag_kxn(kV, 16, lane & 15, lane, 0), c);
      c = wmma16(a_frag_f16(w16 + job * 16 * 64, 64, lane, 32),
                 b_frag_kxn(kV, 16, lane & 15, lane, 32), c);
      int col = h * 16 + (lane & 15);
#pragma unroll
      for (int v = 0; v < 8; ++v)
        ao16[(job * 16 + r0 + v) * 64 + col] = (half_t)c[v];
    }
    __syncthreads();
  }

  // averaged attention weights -> d_out (B,T,L)
  for (int i = tid; i < TS * 64; i += 256) attnw[b * TS * 64 + i] = wsum[i];

  // attn_out = ao (48x64) @ Wo^T + bo -> po16
  for (int job = wave; job < 12; job += 8) {
    int mt = job >> 2, nt = job & 3;
    int nc = nt * 16 + (lane & 15);
    v8f c = {0.f, 0.f, 0.f, 0.f, 0.f, 0.f, 0.f, 0.f};
    c = wmma16(a_frag_f16(ao16 + mt * 16 * 64, 64, lane, 0),
               b_frag_w32(Wo, HH, nc, lane, 0), c);
    c = wmma16(a_frag_f16(ao16 + mt * 16 * 64, 64, lane, 32),
               b_frag_w32(Wo, HH, nc, lane, 32), c);
    float bb = bo[nc];
#pragma unroll
    for (int v = 0; v < 8; ++v)
      po16[(mt * 16 + r0 + v) * 64 + nc] = (half_t)(c[v] + bb);
  }
  __syncthreads();

  // FC1: relu(po @ W1^T + b1) -> 48x32
  for (int job = wave; job < 6; job += 8) {
    int mt = job >> 1, nt = job & 1;
    int nc = nt * 16 + (lane & 15);
    v8f c = {0.f, 0.f, 0.f, 0.f, 0.f, 0.f, 0.f, 0.f};
    c = wmma16(a_frag_f16(po16 + mt * 16 * 64, 64, lane, 0),
               b_frag_w32(W1, HH, nc, lane, 0), c);
    c = wmma16(a_frag_f16(po16 + mt * 16 * 64, 64, lane, 32),
               b_frag_w32(W1, HH, nc, lane, 32), c);
    float bb = b1[nc];
#pragma unroll
    for (int v = 0; v < 8; ++v)
      h16[(mt * 16 + r0 + v) * 32 + nc] = (half_t)fmaxf(c[v] + bb, 0.f);
  }
  __syncthreads();

  // FC2: (48x32) @ W2^T (3x32) + b2 -> pred (B,T,3)
  if (tid < TS * 3) {
    int t = tid / 3, n = tid % 3;
    float acc = b2[n];
    for (int k = 0; k < 32; ++k) acc += (float)h16[t * 32 + k] * W2[n * 32 + k];
    pred[b * TS * 3 + tid] = acc;
  }
}

// ---------------------------------------------------------------------------
extern "C" void kernel_launch(void* const* d_in, const int* in_sizes, int n_in,
                              void* d_out, int out_size, void* d_ws, size_t ws_size,
                              hipStream_t stream)
{
  (void)in_sizes; (void)n_in; (void)out_size; (void)ws_size;

  const float* x        = (const float*)d_in[0];
  const float* target   = (const float*)d_in[1];
  const float* eWih0 = (const float*)d_in[2];
  const float* eWhh0 = (const float*)d_in[3];
  const float* ebih0 = (const float*)d_in[4];
  const float* ebhh0 = (const float*)d_in[5];
  const float* eWih1 = (const float*)d_in[6];
  const float* eWhh1 = (const float*)d_in[7];
  const float* ebih1 = (const float*)d_in[8];
  const float* ebhh1 = (const float*)d_in[9];
  const float* dWih0 = (const float*)d_in[10];
  const float* dWhh0 = (const float*)d_in[11];
  const float* dbih0 = (const float*)d_in[12];
  const float* dbhh0 = (const float*)d_in[13];
  const float* dWih1 = (const float*)d_in[14];
  const float* dWhh1 = (const float*)d_in[15];
  const float* dbih1 = (const float*)d_in[16];
  const float* dbhh1 = (const float*)d_in[17];
  const float* Wqkv  = (const float*)d_in[18];
  const float* bqkv  = (const float*)d_in[19];
  const float* Wo    = (const float*)d_in[20];
  const float* bo    = (const float*)d_in[21];
  const float* W1    = (const float*)d_in[22];
  const float* b1    = (const float*)d_in[23];
  const float* W2    = (const float*)d_in[24];
  const float* b2    = (const float*)d_in[25];

  // workspace carving
  size_t off = 0;
  auto carve = [&](size_t bytes) -> void* {
    void* p = (char*)d_ws + off;
    off += (bytes + 255) & ~(size_t)255;
    return p;
  };
  half_t* enc_h0  = (half_t*)carve((size_t)NB * LS * HH * sizeof(half_t));
  half_t* enc_out = (half_t*)carve((size_t)NB * LS * HH * sizeof(half_t));
  half_t* kv16    = (half_t*)carve((size_t)NB * LS * 128 * sizeof(half_t));
  half_t* dec_h0  = (half_t*)carve((size_t)NB * TS * HH * sizeof(half_t));
  half_t* dec_o16 = (half_t*)carve((size_t)NB * TS * HH * sizeof(half_t));
  float*  hT0     = (float*)carve((size_t)NB * HH * sizeof(float));
  float*  cT0     = (float*)carve((size_t)NB * HH * sizeof(float));
  float*  hT1     = (float*)carve((size_t)NB * HH * sizeof(float));
  float*  cT1     = (float*)carve((size_t)NB * HH * sizeof(float));
  float*  hTd     = (float*)carve((size_t)NB * HH * sizeof(float));
  float*  cTd     = (float*)carve((size_t)NB * HH * sizeof(float));
  float*  dec_in0 = (float*)carve((size_t)NB * HH * sizeof(float));

  float* pred  = (float*)d_out;
  float* attnw = pred + (size_t)NB * TS * 3;

  // 1) encoder layer 0 (zero initial state)
  lstm_scan_kernel<<<NB / 16, 512, 0, stream>>>(
      0, LS, x, nullptr, nullptr, eWih0, eWhh0, ebih0, ebhh0,
      nullptr, nullptr, enc_h0, hT0, cT0);
  // 2) encoder layer 1
  lstm_scan_kernel<<<NB / 16, 512, 0, stream>>>(
      1, LS, nullptr, enc_h0, nullptr, eWih1, eWhh1, ebih1, ebhh1,
      nullptr, nullptr, enc_out, hT1, cT1);
  // 3) K/V projections of encoder outputs
  kv_proj_kernel<<<(NB * LS) / 16, 256, 0, stream>>>(enc_out, Wqkv, bqkv, kv16);
  // 4) step-0 self-attention at the last encoder position -> decoder input 0
  attn0_kernel<<<NB, 128, 0, stream>>>(enc_out, kv16, Wqkv, bqkv, Wo, bo, dec_in0);
  // 5) decoder layer 0 (init from encoder layer-0 finals; teacher forcing)
  lstm_scan_kernel<<<NB / 16, 512, 0, stream>>>(
      2, TS, target, nullptr, dec_in0, dWih0, dWhh0, dbih0, dbhh0,
      hT0, cT0, dec_h0, hTd, cTd);
  // 6) decoder layer 1 (init from encoder layer-1 finals)
  lstm_scan_kernel<<<NB / 16, 512, 0, stream>>>(
      1, TS, nullptr, dec_h0, nullptr, dWih1, dWhh1, dbih1, dbhh1,
      hT1, cT1, dec_o16, hTd, cTd);
  // 7) cross attention + Wo + FC head, writes both outputs
  cross_attn_kernel<<<NB, 256, 0, stream>>>(
      dec_o16, kv16, Wqkv, bqkv, Wo, bo, W1, b1, W2, b2, pred, attnw);
}
